// SpatialODE_52286931861959
// MI455X (gfx1250) — compile-verified, matrix operand order checked by reading, not run on hardware
//
#include <hip/hip_runtime.h>
#include <hip/hip_bf16.h>

// ---------------- problem constants ----------------
constexpr int Db   = 64;           // feature dim D
constexpr int DM   = 320;          // m*D = 5*64
constexpr int FFd  = 256;          // hidden FF
constexpr int NTOK = 32 * 12 * 2048;
constexpr int NTILES = NTOK / 16;  // 49152 16-token tiles
constexpr int NWAVE = 8;           // waves per block (256 threads)
constexpr int NBLOCKS = 512;
constexpr float HSTEP = 0.25f;
constexpr float LNEPS = 1e-5f;

// LDS strides (bf16 elements); padded so row stride is 16B-aligned and
// bank-conflict friendly (stride*2 bytes ≡ 656B -> 36 dwords mod 64 per row)
constexpr int HSTR = 328;   // hist  [16][328]   (cat: 5 states * 64)
constexpr int FSTR = 72;    // ffstg [16][72]    (64-wide FF chunk)
constexpr int WS1  = 328;   // W1t   [256][328]  (K=320)
constexpr int WS2  = 264;   // W2t   [64][264]   (K=256)
constexpr int WSO  = 72;    // Wot   [64][72]    (K=64)

// shared memory layout (offsets in bf16 units)
constexpr size_t OFF_W1T  = 0;
constexpr size_t OFF_W2T  = OFF_W1T  + (size_t)256 * WS1;
constexpr size_t OFF_WOT  = OFF_W2T  + (size_t)64  * WS2;
constexpr size_t OFF_HIST = OFF_WOT  + (size_t)64  * WSO;
constexpr size_t OFF_FFS  = OFF_HIST + (size_t)NWAVE * 16 * HSTR;
constexpr size_t OFF_F32  = OFF_FFS  + (size_t)NWAVE * 16 * FSTR; // float region
constexpr size_t SHMEM_BYTES = OFF_F32 * 2 + 512 * 4;             // + biases/gamma/beta

typedef __bf16 bf16_t;
typedef __attribute__((ext_vector_type(16))) __bf16 v16bf;
typedef __attribute__((ext_vector_type(8)))  __bf16 v8bf;
typedef __attribute__((ext_vector_type(8)))  float  v8f;

__device__ __forceinline__ void wave_sync() {
  __builtin_amdgcn_wave_barrier();
  asm volatile("" ::: "memory");
}

// Load one 16x32 bf16 A/B operand fragment from row-major LDS [row][K],
// row stride `stride` (bf16 units, 16B aligned). 16-bit operand layout:
// lane L: r=L%16, h=L/16; dwords 0..3 = K(8h .. 8h+7), dwords 4..7 = K(16+8h ..).
__device__ __forceinline__ v16bf load_frag(const bf16_t* base, int stride,
                                           int row0, int col0, int lane) {
  const int r = lane & 15, h = lane >> 4;
  const bf16_t* p = base + (size_t)(row0 + r) * stride + col0 + 8 * h;
  v8bf lo = *(const v8bf*)(p);
  v8bf hi = *(const v8bf*)(p + 16);
  union { struct { v8bf a, b; } s; v16bf v; } u;
  u.s.a = lo; u.s.b = hi;
  return u.v;
}

// Scatter a C-layout v8f accumulator tile (16 rows x 16 cols at col0) into a
// row-major bf16 LDS matrix. C layout: VGPR v, lane L -> (M = v + 8*(L/16), N = L%16).
__device__ __forceinline__ void store_tile_bf16(bf16_t* base, int stride,
                                                int col0, int lane, v8f t) {
  const int nn = lane & 15, h = lane >> 4;
#pragma unroll
  for (int v = 0; v < 8; ++v)
    base[(size_t)(v + 8 * h) * stride + col0 + nn] = (bf16_t)t[v];
}

__device__ __forceinline__ v8f wmma_bf16(v16bf a, v16bf b, v8f c) {
  return __builtin_amdgcn_wmma_f32_16x16x32_bf16(
      /*neg_a=*/false, a, /*neg_b=*/false, b,
      /*c_mod=*/(short)0, c, /*reuse_a=*/false, /*reuse_b=*/false);
}

__global__ __launch_bounds__(256, 1)
void SpatialODE_52286931861959_kernel(const float* __restrict__ x,
                                      const float* __restrict__ W_ode,
                                      const float* __restrict__ b_ode,
                                      const float* __restrict__ W1,
                                      const float* __restrict__ b1,
                                      const float* __restrict__ W2,
                                      const float* __restrict__ b2,
                                      const float* __restrict__ gamma,
                                      const float* __restrict__ beta,
                                      float* __restrict__ out) {
  extern __shared__ char smem_raw[];
  bf16_t* sm   = (bf16_t*)smem_raw;
  bf16_t* W1t  = sm + OFF_W1T;   // [256][WS1] = W1^T  (row = ff out, col = K in 0..319)
  bf16_t* W2t  = sm + OFF_W2T;   // [64][WS2]  = W2^T
  bf16_t* Wot  = sm + OFF_WOT;   // [64][WSO]  = W_ode^T
  bf16_t* histA = sm + OFF_HIST; // per-wave [16][HSTR]
  bf16_t* ffsA  = sm + OFF_FFS;  // per-wave [16][FSTR]
  float*  sb_ode = (float*)(sm + OFF_F32);
  float*  sb1    = sb_ode + 64;
  float*  sb2    = sb1 + 256;
  float*  sgam   = sb2 + 64;
  float*  sbet   = sgam + 64;

  const int tid = threadIdx.x;

  // ---- stage weights (bf16, transposed) once per workgroup ----
  for (int i = tid; i < 64 * 64; i += 256) {       // W_ode [d][e] -> Wot[e][d]
    int k = i >> 6, n = i & 63;
    Wot[(size_t)n * WSO + k] = (bf16_t)W_ode[i];
  }
  for (int i = tid; i < DM * FFd; i += 256) {      // W1 [m][f] -> W1t[f][m]
    int k = i / FFd, n = i % FFd;
    W1t[(size_t)n * WS1 + k] = (bf16_t)W1[i];
  }
  for (int i = tid; i < FFd * Db; i += 256) {      // W2 [f][d] -> W2t[d][f]
    int k = i >> 6, n = i & 63;
    W2t[(size_t)n * WS2 + k] = (bf16_t)W2[i];
  }
  if (tid < 64) {
    sb_ode[tid] = b_ode[tid];
    sb2[tid]    = b2[tid];
    sgam[tid]   = gamma[tid];
    sbet[tid]   = beta[tid];
  }
  sb1[tid] = b1[tid];  // 256 threads, 256 entries
  __syncthreads();

  const int wave = tid >> 5, lane = tid & 31;
  const int nn = lane & 15, h = lane >> 4;
  bf16_t* hist = histA + (size_t)wave * 16 * HSTR;
  bf16_t* ffs  = ffsA  + (size_t)wave * 16 * FSTR;

  const int gwave   = blockIdx.x * NWAVE + wave;
  const int gstride = gridDim.x * NWAVE;

  for (int tile = gwave; tile < NTILES; tile += gstride) {
    const long tok0 = (long)tile * 16;
    const float* xt = x + tok0 * Db;

    // prefetch next tile's x (global_prefetch_b8): 32 lanes x 128B = 4KB tile
    if (tile + gstride < NTILES)
      __builtin_prefetch(x + (long)(tile + gstride) * 16 * Db + lane * 32, 0, 1);

    // ---- 1) load x in C layout (fp32 residual + initial state), seed hist s0 ----
    v8f s[4], resid[4];
#pragma unroll
    for (int n = 0; n < 4; ++n) {
      v8f t;
#pragma unroll
      for (int v = 0; v < 8; ++v)
        t[v] = xt[(size_t)(v + 8 * h) * Db + n * 16 + nn];
      resid[n] = t;
      s[n] = t;
      store_tile_bf16(hist, HSTR, n * 16, lane, t);   // state slot 0 = x
    }
    wave_sync();

    // ---- 2) Euler ODE: s += h * tanh(s @ W_ode + b_ode), 4 steps ----
#pragma unroll
    for (int k = 1; k <= 4; ++k) {
      v16bf a0 = load_frag(hist, HSTR, 0, (k - 1) * 64,      lane);
      v16bf a1 = load_frag(hist, HSTR, 0, (k - 1) * 64 + 32, lane);
#pragma unroll
      for (int n = 0; n < 4; ++n) {
        v8f u = {0.f, 0.f, 0.f, 0.f, 0.f, 0.f, 0.f, 0.f};
        u = wmma_bf16(a0, load_frag(Wot, WSO, n * 16, 0,  lane), u);
        u = wmma_bf16(a1, load_frag(Wot, WSO, n * 16, 32, lane), u);
        const float bb = sb_ode[n * 16 + nn];
        v8f t = s[n];
#pragma unroll
        for (int v = 0; v < 8; ++v) t[v] = t[v] + HSTEP * tanhf(u[v] + bb);
        s[n] = t;
        store_tile_bf16(hist, HSTR, k * 64 + n * 16, lane, t);  // state slot k
      }
      wave_sync();
    }

    // ---- 3) proj: relu(cat @ W1 + b1) @ W2, fused over 64-wide FF chunks ----
    v16bf aH[10];
#pragma unroll
    for (int sl = 0; sl < 10; ++sl)
      aH[sl] = load_frag(hist, HSTR, 0, sl * 32, lane);   // cat A operand, K=320

    v8f acc[4];
#pragma unroll
    for (int n = 0; n < 4; ++n) acc[n] = {0.f, 0.f, 0.f, 0.f, 0.f, 0.f, 0.f, 0.f};

#pragma unroll
    for (int c = 0; c < 4; ++c) {              // FF chunk: cols c*64 .. c*64+63
#pragma unroll
      for (int fn = 0; fn < 4; ++fn) {
        const int f = c * 4 + fn;              // 16-wide FF tile index
        v8f u = {0.f, 0.f, 0.f, 0.f, 0.f, 0.f, 0.f, 0.f};
#pragma unroll
        for (int sl = 0; sl < 10; ++sl)
          u = wmma_bf16(aH[sl], load_frag(W1t, WS1, f * 16, sl * 32, lane), u);
        const float bb = sb1[f * 16 + nn];
#pragma unroll
        for (int v = 0; v < 8; ++v) {
          float z = u[v] + bb;
          u[v] = z > 0.f ? z : 0.f;            // ReLU
        }
        store_tile_bf16(ffs, FSTR, fn * 16, lane, u);
      }
      wave_sync();
#pragma unroll
      for (int s2 = 0; s2 < 2; ++s2) {         // K-slices of this FF chunk
        v16bf af = load_frag(ffs, FSTR, 0, s2 * 32, lane);
#pragma unroll
        for (int n = 0; n < 4; ++n)
          acc[n] = wmma_bf16(af, load_frag(W2t, WS2, n * 16, c * 64 + s2 * 32, lane),
                             acc[n]);
      }
      wave_sync();
    }

    // ---- 4) residual + LayerNorm(D=64) + store ----
#pragma unroll
    for (int v = 0; v < 8; ++v) {              // row M = v + 8h
      float zv[4];
      float zsum = 0.f, zsq = 0.f;
#pragma unroll
      for (int n = 0; n < 4; ++n) {
        float z = acc[n][v] + sb2[n * 16 + nn] + resid[n][v];
        zv[n] = z;
        zsum += z;
        zsq  += z * z;
      }
#pragma unroll
      for (int m = 1; m < 16; m <<= 1) {       // butterfly within 16-lane group
        zsum += __shfl_xor(zsum, m, 32);
        zsq  += __shfl_xor(zsq,  m, 32);
      }
      const float mu  = zsum * (1.f / 64.f);
      const float var = zsq * (1.f / 64.f) - mu * mu;
      const float rs  = rsqrtf(var + LNEPS);
      float* op = out + (tok0 + v + 8 * h) * (long)Db;
#pragma unroll
      for (int n = 0; n < 4; ++n) {
        const int col = n * 16 + nn;
        op[col] = (zv[n] - mu) * rs * sgam[col] + sbet[col];
      }
    }
  }
}

extern "C" void kernel_launch(void* const* d_in, const int* in_sizes, int n_in,
                              void* d_out, int out_size, void* d_ws, size_t ws_size,
                              hipStream_t stream) {
  (void)in_sizes; (void)n_in; (void)out_size; (void)d_ws; (void)ws_size;
  const float* x     = (const float*)d_in[0];
  const float* W_ode = (const float*)d_in[1];
  const float* b_ode = (const float*)d_in[2];
  const float* W1    = (const float*)d_in[3];
  const float* b1    = (const float*)d_in[4];
  const float* W2    = (const float*)d_in[5];
  const float* b2    = (const float*)d_in[6];
  const float* gamma = (const float*)d_in[7];
  const float* beta  = (const float*)d_in[8];
  float* out = (float*)d_out;

  SpatialODE_52286931861959_kernel<<<dim3(NBLOCKS), dim3(256), SHMEM_BYTES, stream>>>(
      x, W_ode, b_ode, W1, b1, W2, b2, gamma, beta, out);
}
